// Decoder_48052094107929
// MI455X (gfx1250) — matrix operand order repaired and checked by reading, not compiled, not run on hardware
//
#include <hip/hip_runtime.h>
#include <hip/hip_bf16.h>
#include <math.h>

// ---------------- problem constants ----------------
static constexpr int Bn   = 8192;
static constexpr int DIMn = 1024;
static constexpr int En   = 4;
static constexpr int HDn  = 256;       // DIM / H
static constexpr int HIDn = 4096;      // 4 * DIM
static constexpr float SCALEf = 0.0625f;   // HD^-0.5 = 1/16
static constexpr float EPSf   = 1e-5f;
static constexpr int KCHUNK = 128;     // K elements staged per TDM tile

// ---------------- types ----------------
typedef __bf16 bf16_t;
typedef __attribute__((ext_vector_type(16))) __bf16 bf16x16;
typedef __attribute__((ext_vector_type(8)))  __bf16 bf16x8;
typedef __attribute__((ext_vector_type(8)))  float  f32x8;
typedef __attribute__((ext_vector_type(4)))  unsigned int u32x4;
typedef __attribute__((ext_vector_type(8)))  int i32x8;
typedef __attribute__((ext_vector_type(4)))  int i32x4;

__device__ __forceinline__ bf16_t to_bf16(float f) { return (bf16_t)f; }

// LDS byte offset of a __shared__ object = low 32 bits of its generic address
// (LDS aperture: addr[31:0] is the allocation-relative LDS offset).
__device__ __forceinline__ unsigned lds_off(const void* p) {
    return (unsigned)(uintptr_t)p;
}

// ---------------- WMMA fragment loads (CDNA5 ISA 7.12.2 layouts) ----------------
// A: 16x32 bf16. lane L: m = L&15, h = L>>4.
//   halves 0..7  = A[row][k0 + 8h .. +7] ; halves 8..15 = A[row][k0 + 16 + 8h .. +7]
__device__ __forceinline__ bf16x16 load_afrag(const bf16_t* __restrict__ A,
                                              int Kdim, int rowAbs, int k0, int h) {
    union { bf16x16 v; bf16x8 q[2]; } u;
    const bf16_t* p = A + (size_t)rowAbs * Kdim + k0 + h * 8;
    u.q[0] = *(const bf16x8*)(p);
    u.q[1] = *(const bf16x8*)(p + 16);
    return u.v;
}
// B fragment from an LDS-staged tile laid out [n][KCHUNK] (k contiguous):
// lane L: n = L&15, h = L>>4 -> halves 0..7 at k = kk + 8h, halves 8..15 at k = kk + 16 + 8h.
__device__ __forceinline__ bf16x16 load_bfrag_lds(const bf16_t* __restrict__ Bt,
                                                  int nrel, int kk, int h) {
    union { bf16x16 v; bf16x8 q[2]; } u;
    const bf16_t* p = Bt + (size_t)nrel * KCHUNK + kk + h * 8;
    u.q[0] = *(const bf16x8*)(p);
    u.q[1] = *(const bf16x8*)(p + 16);
    return u.v;
}

__device__ __forceinline__ f32x8 wmma_bf16(bf16x16 a, bf16x16 b, f32x8 c) {
    return __builtin_amdgcn_wmma_f32_16x16x32_bf16(false, a, false, b, (short)0, c,
                                                   false, false);
}

// ---------------- TDM: issue a 2D weight-tile load (64 N-rows x KCHUNK) --------
// D# group0/group1 packed per CDNA5 ISA §8.3/§8.4. Groups 2/3 zero (2D tile).
__device__ __forceinline__ void tdm_issue_b(const bf16_t* W, int Kdim, int Ntot,
                                            int col0, int k0, unsigned lds_addr) {
    unsigned long long ga = (unsigned long long)(uintptr_t)
                            (W + (size_t)col0 * Kdim + k0);
    u32x4 g0;
    g0[0] = 1u;                                   // count=1, user mode
    g0[1] = lds_addr;                             // lds_addr
    g0[2] = (unsigned)ga;                         // global_addr[31:0]
    g0[3] = (unsigned)((ga >> 32) & 0x01FFFFFFu)  // global_addr[56:32]
            | (2u << 30);                         // type = 2 (image)
    i32x8 g1;
    g1[0] = 0x10000;                              // data_size=1 (2 bytes/elem)
    g1[1] = (int)(((unsigned)Kdim & 0xFFFFu) << 16);            // tensor_dim0 lo
    g1[2] = (int)((((unsigned)Kdim >> 16) & 0xFFFFu)
            | (((unsigned)Ntot & 0xFFFFu) << 16));              // dim0 hi | dim1 lo
    g1[3] = (int)((((unsigned)Ntot >> 16) & 0xFFFFu)
            | ((unsigned)KCHUNK << 16));                        // dim1 hi | tile_dim0
    g1[4] = 64;                                   // tile_dim1=64, tile_dim2=0
    g1[5] = Kdim;                                 // tensor_dim0_stride[31:0]
    g1[6] = 0;
    g1[7] = 0;
    i32x4 gz = {0, 0, 0, 0};
    asm volatile("tensor_load_to_lds %0, %1, %2, %3"
                 :: "s"(g0), "s"(g1), "s"(gz), "s"(gz)
                 : "memory");
}

// ---------------- staged block GEMM: acc[c] += A @ W^T over full K ------------
// Block = 8 waves stacked in M; B tile (64 cols) shared via LDS double buffer.
__device__ __forceinline__ void gemm_staged(const bf16_t* __restrict__ A,
                                            const bf16_t* __restrict__ W,
                                            int Kdim, int Ntot,
                                            bf16_t* lds0, bf16_t* lds1,
                                            int row0, int col0,
                                            int wave, int l15, int h,
                                            f32x8 acc[4]) {
    const int nch = Kdim / KCHUNK;
    if (wave == 0) {
        tdm_issue_b(W, Kdim, Ntot, col0, 0, lds_off(lds0));
        if (nch > 1) tdm_issue_b(W, Kdim, Ntot, col0, KCHUNK, lds_off(lds1));
    }
    for (int ch = 0; ch < nch; ++ch) {
        if (wave == 0) {
            if (ch + 1 < nch) __builtin_amdgcn_s_wait_tensorcnt(1);
            else              __builtin_amdgcn_s_wait_tensorcnt(0);
        }
        __syncthreads();
        const bf16_t* Bt = (ch & 1) ? lds1 : lds0;
#pragma unroll
        for (int kk = 0; kk < KCHUNK; kk += 32) {
            bf16x16 a = load_afrag(A, Kdim, row0 + l15, ch * KCHUNK + kk, h);
#pragma unroll
            for (int c = 0; c < 4; ++c) {
                bf16x16 b = load_bfrag_lds(Bt, c * 16 + l15, kk, h);
                acc[c] = wmma_bf16(a, b, acc[c]);
            }
        }
        __syncthreads();
        if (wave == 0 && ch + 2 < nch)
            tdm_issue_b(W, Kdim, Ntot, col0, (ch + 2) * KCHUNK,
                        lds_off((ch & 1) ? lds1 : lds0));
    }
}

// ---------------- kernel 0a: f32 -> bf16 convert (grid-stride) ----------------
__global__ void cvt_bf16_kernel(const float* __restrict__ src,
                                bf16_t* __restrict__ dst, size_t n) {
    size_t i = (size_t)blockIdx.x * blockDim.x + threadIdx.x;
    size_t stride = (size_t)gridDim.x * blockDim.x;
    for (; i < n; i += stride) dst[i] = to_bf16(src[i]);
}

// ---------------- kernel 0b: split Wqkv -> Wq (bf16) and Wk+Wv (bf16) ---------
__global__ void build_qkv_kernel(const float* __restrict__ Wqkv,
                                 bf16_t* __restrict__ wq,
                                 bf16_t* __restrict__ wkv) {
    size_t n = (size_t)En * DIMn * DIMn;
    size_t i = (size_t)blockIdx.x * blockDim.x + threadIdx.x;
    size_t stride = (size_t)gridDim.x * blockDim.x;
    for (; i < n; i += stride) {
        size_t e   = i / ((size_t)DIMn * DIMn);
        size_t rem = i % ((size_t)DIMn * DIMn);
        size_t f   = rem / DIMn;
        size_t d   = rem % DIMn;
        const float* base = Wqkv + e * (size_t)3 * DIMn * DIMn;
        wq[i]  = to_bf16(base[f * DIMn + d]);
        wkv[i] = to_bf16(base[((size_t)DIMn + f) * DIMn + d] +
                         base[((size_t)2 * DIMn + f) * DIMn + d]);
    }
}

// ---------------- kernel 1: per-row prep (layernorms, x->bf16, gate top-2) ----
__global__ void prep_kernel(const float* __restrict__ x, const float* __restrict__ y,
                            const float* __restrict__ Wg, const float* __restrict__ bg,
                            const float* __restrict__ g1, const float* __restrict__ bn1,
                            const float* __restrict__ g2, const float* __restrict__ bn2,
                            bf16_t* __restrict__ xb, bf16_t* __restrict__ yn1,
                            bf16_t* __restrict__ yn2, float* __restrict__ cbuf) {
    __shared__ float red[256];
    __shared__ float sstat[2];
    int row = blockIdx.x;
    int t   = threadIdx.x;
    const float* yr = y + (size_t)row * DIMn;
    const float* xr = x + (size_t)row * DIMn;

    float yv[4], xv[4];
    float s = 0.f, s2 = 0.f;
#pragma unroll
    for (int j = 0; j < 4; ++j) {
        int col = t + j * 256;
        yv[j] = yr[col];
        xv[j] = xr[col];
        s  += yv[j];
        s2 += yv[j] * yv[j];
    }
    red[t] = s;  __syncthreads();
    for (int st = 128; st > 0; st >>= 1) { if (t < st) red[t] += red[t + st]; __syncthreads(); }
    if (t == 0) sstat[0] = red[0] * (1.0f / DIMn);
    __syncthreads();
    red[t] = s2; __syncthreads();
    for (int st = 128; st > 0; st >>= 1) { if (t < st) red[t] += red[t + st]; __syncthreads(); }
    if (t == 0) sstat[1] = red[0] * (1.0f / DIMn);
    __syncthreads();

    float mean = sstat[0];
    float var  = sstat[1] - mean * mean;
    float rs   = rsqrtf(var + EPSf);

#pragma unroll
    for (int j = 0; j < 4; ++j) {
        int col = t + j * 256;
        float nv = (yv[j] - mean) * rs;
        size_t idx = (size_t)row * DIMn + col;
        yn1[idx] = to_bf16(nv * g1[col] + bn1[col]);
        yn2[idx] = to_bf16(nv * g2[col] + bn2[col]);
        xb[idx]  = to_bf16(xv[j]);
    }

    float gp[4] = {0.f, 0.f, 0.f, 0.f};
#pragma unroll
    for (int j = 0; j < 4; ++j) {
        int col = t + j * 256;
#pragma unroll
        for (int e = 0; e < 4; ++e) gp[e] += xv[j] * Wg[(size_t)e * DIMn + col];
    }
    __shared__ float sl[4];
    for (int e = 0; e < 4; ++e) {
        __syncthreads();
        red[t] = gp[e]; __syncthreads();
        for (int st = 128; st > 0; st >>= 1) { if (t < st) red[t] += red[t + st]; __syncthreads(); }
        if (t == 0) sl[e] = red[0] + bg[e];
    }
    __syncthreads();
    if (t == 0) {
        float mx = sl[0];
        for (int e = 1; e < 4; ++e) mx = fmaxf(mx, sl[e]);
        float p[4], sum = 0.f;
        for (int e = 0; e < 4; ++e) { p[e] = __expf(sl[e] - mx); sum += p[e]; }
        float inv = 1.0f / sum;
        for (int e = 0; e < 4; ++e) p[e] *= inv;
        int i1 = 0;
        for (int e = 1; e < 4; ++e) if (p[e] > p[i1]) i1 = e;
        int i2 = -1;
        for (int e = 0; e < 4; ++e) if (e != i1 && (i2 < 0 || p[e] > p[i2])) i2 = e;
        for (int e = 0; e < 4; ++e)
            cbuf[(size_t)row * En + e] = (e == i1 || e == i2) ? p[e] : 0.f;
    }
}

// ---------------- kernel 2: MoE dual-GEMM with TDM-staged weight tiles --------
// s = sum_e c_e * (yn1@Wq[e]^T + x@Wkv[e]^T). 32 pipelined steps = 4 experts x 8 chunks.
__global__ void __launch_bounds__(256)
moe_gemm_kernel(const bf16_t* __restrict__ yn1, const bf16_t* __restrict__ xb,
                const bf16_t* __restrict__ wq, const bf16_t* __restrict__ wkv,
                const float* __restrict__ cbuf, float* __restrict__ sbuf) {
    __shared__ bf16_t lbq[2][64 * KCHUNK];
    __shared__ bf16_t lbk[2][64 * KCHUNK];
    int wave = threadIdx.x >> 5;
    int lane = threadIdx.x & 31;
    int l15  = lane & 15;
    int h    = lane >> 4;
    int row0 = blockIdx.y * 128 + wave * 16;
    int col0 = blockIdx.x * 64;

    const int NCH   = DIMn / KCHUNK;   // 8 chunks per expert
    const int STEPS = En * NCH;        // 32

    if (wave == 0) {
        // prime two steps (a step = one Wq tile + one Wkv tile)
        for (int s = 0; s < 2; ++s) {
            int e = s >> 3, ch = s & 7, buf = s & 1;
            tdm_issue_b(wq  + (size_t)e * DIMn * DIMn, DIMn, DIMn, col0,
                        ch * KCHUNK, lds_off(lbq[buf]));
            tdm_issue_b(wkv + (size_t)e * DIMn * DIMn, DIMn, DIMn, col0,
                        ch * KCHUNK, lds_off(lbk[buf]));
        }
    }

    f32x8 fin[4] = {};
    f32x8 acc[4] = {};
    for (int step = 0; step < STEPS; ++step) {
        int e = step >> 3, ch = step & 7, buf = step & 1;
        if (wave == 0) {
            if (step + 1 < STEPS) __builtin_amdgcn_s_wait_tensorcnt(2);
            else                  __builtin_amdgcn_s_wait_tensorcnt(0);
        }
        __syncthreads();
#pragma unroll
        for (int kk = 0; kk < KCHUNK; kk += 32) {
            int k0 = ch * KCHUNK + kk;
            bf16x16 ay = load_afrag(yn1, DIMn, row0 + l15, k0, h);
#pragma unroll
            for (int c = 0; c < 4; ++c) {
                bf16x16 b = load_bfrag_lds(lbq[buf], c * 16 + l15, kk, h);
                acc[c] = wmma_bf16(ay, b, acc[c]);
            }
            bf16x16 ax = load_afrag(xb, DIMn, row0 + l15, k0, h);
#pragma unroll
            for (int c = 0; c < 4; ++c) {
                bf16x16 b = load_bfrag_lds(lbk[buf], c * 16 + l15, kk, h);
                acc[c] = wmma_bf16(ax, b, acc[c]);
            }
        }
        __syncthreads();
        if (wave == 0 && step + 2 < STEPS) {
            int s2 = step + 2;
            int e2 = s2 >> 3, ch2 = s2 & 7, buf2 = s2 & 1;
            tdm_issue_b(wq  + (size_t)e2 * DIMn * DIMn, DIMn, DIMn, col0,
                        ch2 * KCHUNK, lds_off(lbq[buf2]));
            tdm_issue_b(wkv + (size_t)e2 * DIMn * DIMn, DIMn, DIMn, col0,
                        ch2 * KCHUNK, lds_off(lbk[buf2]));
        }
        if (ch == NCH - 1) {   // finished expert e: apply per-row gate coefficient
#pragma unroll
            for (int r = 0; r < 8; ++r) {
                float cv = cbuf[(size_t)(row0 + h * 8 + r) * En + e];
#pragma unroll
                for (int c = 0; c < 4; ++c) fin[c][r] += cv * acc[c][r];
            }
#pragma unroll
            for (int c = 0; c < 4; ++c) acc[c] = (f32x8)0.0f;
        }
    }
#pragma unroll
    for (int c = 0; c < 4; ++c)
#pragma unroll
        for (int r = 0; r < 8; ++r)
            sbuf[(size_t)(row0 + h * 8 + r) * DIMn + col0 + c * 16 + l15] = fin[c][r];
}

// ---------------- kernel 3: per-row 4x4 head attention ----------------
__global__ void attn_kernel(const float* __restrict__ sbuf, bf16_t* __restrict__ abuf) {
    __shared__ float sh[1024];
    __shared__ float red[16][17];
    __shared__ float aw[16];
    int row = blockIdx.x;
    int t   = threadIdx.x;
    const float* sr = sbuf + (size_t)row * DIMn;
    for (int j = t; j < 1024; j += 256) sh[j] = sr[j];
    __syncthreads();

    int p = t >> 4, l = t & 15;
    int hh = p >> 2, gg = p & 3;
    float part = 0.f;
    for (int i = l; i < HDn; i += 16) part += sh[hh * HDn + i] * sh[gg * HDn + i];
    red[p][l] = part;
    __syncthreads();
    if (t < 16) {
        float s = 0.f;
#pragma unroll
        for (int i = 0; i < 16; ++i) s += red[t][i];
        aw[t] = s * SCALEf;
    }
    __syncthreads();
    if (t < 4) {
        float m = aw[t * 4];
        for (int g = 1; g < 4; ++g) m = fmaxf(m, aw[t * 4 + g]);
        float ex[4], sum = 0.f;
        for (int g = 0; g < 4; ++g) { ex[g] = __expf(aw[t * 4 + g] - m); sum += ex[g]; }
        float inv = 1.0f / sum;
        for (int g = 0; g < 4; ++g) aw[t * 4 + g] = ex[g] * inv;
    }
    __syncthreads();
    for (int idx = t; idx < 1024; idx += 256) {
        int ho = idx >> 8, d = idx & 255;
        float o = 0.f;
#pragma unroll
        for (int g = 0; g < 4; ++g) o += aw[ho * 4 + g] * sh[g * HDn + d];
        abuf[(size_t)row * DIMn + idx] = to_bf16(o);
    }
}

// ---------------- kernel 4: out = y + attn_out @ Wp^T + bp ----------------
__global__ void __launch_bounds__(256)
gemm_wp_kernel(const bf16_t* __restrict__ abuf, const bf16_t* __restrict__ wp,
               const float* __restrict__ y, const float* __restrict__ bp,
               float* __restrict__ outp) {
    __shared__ bf16_t lb0[64 * KCHUNK];
    __shared__ bf16_t lb1[64 * KCHUNK];
    int wave = threadIdx.x >> 5;
    int lane = threadIdx.x & 31;
    int l15  = lane & 15;
    int h    = lane >> 4;
    int row0 = blockIdx.y * 128 + wave * 16;
    int col0 = blockIdx.x * 64;
    f32x8 acc[4] = {};
    gemm_staged(abuf, wp, DIMn, DIMn, lb0, lb1, row0, col0, wave, l15, h, acc);
#pragma unroll
    for (int c = 0; c < 4; ++c)
#pragma unroll
        for (int r = 0; r < 8; ++r) {
            int col = col0 + c * 16 + l15;
            size_t idx = (size_t)(row0 + h * 8 + r) * DIMn + col;
            outp[idx] = y[idx] + acc[c][r] + bp[col];
        }
}

// ---------------- kernel 5: h = gelu(yn2 @ W1^T + bm1) -> bf16 ----------------
__global__ void __launch_bounds__(256)
gemm_mlp1_kernel(const bf16_t* __restrict__ yn2, const bf16_t* __restrict__ w1,
                 const float* __restrict__ bm1, bf16_t* __restrict__ hbuf) {
    __shared__ bf16_t lb0[64 * KCHUNK];
    __shared__ bf16_t lb1[64 * KCHUNK];
    int wave = threadIdx.x >> 5;
    int lane = threadIdx.x & 31;
    int l15  = lane & 15;
    int h    = lane >> 4;
    int row0 = blockIdx.y * 128 + wave * 16;
    int col0 = blockIdx.x * 64;
    f32x8 acc[4] = {};
    gemm_staged(yn2, w1, DIMn, HIDn, lb0, lb1, row0, col0, wave, l15, h, acc);
#pragma unroll
    for (int c = 0; c < 4; ++c)
#pragma unroll
        for (int r = 0; r < 8; ++r) {
            int col = col0 + c * 16 + l15;
            float v = acc[c][r] + bm1[col];
            v = 0.5f * v * (1.0f + erff(v * 0.70710678118654752f));  // exact gelu
            hbuf[(size_t)(row0 + h * 8 + r) * HIDn + col] = to_bf16(v);
        }
}

// ---------------- kernel 6: out += h @ W2^T + bm2 ----------------
__global__ void __launch_bounds__(256)
gemm_mlp2_kernel(const bf16_t* __restrict__ hbuf, const bf16_t* __restrict__ w2,
                 const float* __restrict__ bm2, float* __restrict__ outp) {
    __shared__ bf16_t lb0[64 * KCHUNK];
    __shared__ bf16_t lb1[64 * KCHUNK];
    int wave = threadIdx.x >> 5;
    int lane = threadIdx.x & 31;
    int l15  = lane & 15;
    int h    = lane >> 4;
    int row0 = blockIdx.y * 128 + wave * 16;
    int col0 = blockIdx.x * 64;
    f32x8 acc[4] = {};
    gemm_staged(hbuf, w2, HIDn, DIMn, lb0, lb1, row0, col0, wave, l15, h, acc);
#pragma unroll
    for (int c = 0; c < 4; ++c)
#pragma unroll
        for (int r = 0; r < 8; ++r) {
            int col = col0 + c * 16 + l15;
            size_t idx = (size_t)(row0 + h * 8 + r) * DIMn + col;
            outp[idx] = outp[idx] + acc[c][r] + bm2[col];
        }
}

// ---------------- host launcher ----------------
extern "C" void kernel_launch(void* const* d_in, const int* in_sizes, int n_in,
                              void* d_out, int out_size, void* d_ws, size_t ws_size,
                              hipStream_t stream) {
    (void)in_sizes; (void)n_in; (void)out_size; (void)ws_size;
    const float* x    = (const float*)d_in[0];
    const float* y    = (const float*)d_in[1];
    const float* Wg   = (const float*)d_in[2];
    const float* bg   = (const float*)d_in[3];
    const float* Wqkv = (const float*)d_in[4];
    const float* Wp   = (const float*)d_in[5];
    const float* bp   = (const float*)d_in[6];
    const float* g1   = (const float*)d_in[7];
    const float* bn1  = (const float*)d_in[8];
    const float* g2   = (const float*)d_in[9];
    const float* bn2  = (const float*)d_in[10];
    const float* W1   = (const float*)d_in[11];
    const float* bm1  = (const float*)d_in[12];
    const float* W2   = (const float*)d_in[13];
    const float* bm2  = (const float*)d_in[14];
    float* outp = (float*)d_out;

    char* w = (char*)d_ws;
    size_t off = 0;
    auto alloc = [&](size_t bytes) {
        void* p = w + off;
        off += (bytes + 255) & ~(size_t)255;
        return p;
    };
    bf16_t* xb   = (bf16_t*)alloc((size_t)Bn * DIMn * 2);
    bf16_t* yn1  = (bf16_t*)alloc((size_t)Bn * DIMn * 2);
    bf16_t* yn2  = (bf16_t*)alloc((size_t)Bn * DIMn * 2);
    bf16_t* wq   = (bf16_t*)alloc((size_t)En * DIMn * DIMn * 2);
    bf16_t* wkv  = (bf16_t*)alloc((size_t)En * DIMn * DIMn * 2);
    bf16_t* wpb  = (bf16_t*)alloc((size_t)DIMn * DIMn * 2);
    bf16_t* w1b  = (bf16_t*)alloc((size_t)HIDn * DIMn * 2);
    bf16_t* w2b  = (bf16_t*)alloc((size_t)DIMn * HIDn * 2);
    float*  cbuf = (float*)alloc((size_t)Bn * En * 4);
    float*  sbuf = (float*)alloc((size_t)Bn * DIMn * 4);
    bf16_t* abuf = (bf16_t*)alloc((size_t)Bn * DIMn * 2);
    bf16_t* hbuf = (bf16_t*)alloc((size_t)Bn * HIDn * 2);

    cvt_bf16_kernel<<<2048, 256, 0, stream>>>(Wp, wpb, (size_t)DIMn * DIMn);
    cvt_bf16_kernel<<<4096, 256, 0, stream>>>(W1, w1b, (size_t)HIDn * DIMn);
    cvt_bf16_kernel<<<4096, 256, 0, stream>>>(W2, w2b, (size_t)DIMn * HIDn);
    build_qkv_kernel<<<4096, 256, 0, stream>>>(Wqkv, wq, wkv);

    prep_kernel<<<Bn, 256, 0, stream>>>(x, y, Wg, bg, g1, bn1, g2, bn2,
                                        xb, yn1, yn2, cbuf);

    moe_gemm_kernel<<<dim3(DIMn / 64, Bn / 128), 256, 0, stream>>>(yn1, xb, wq, wkv,
                                                                   cbuf, sbuf);
    attn_kernel<<<Bn, 256, 0, stream>>>(sbuf, abuf);

    gemm_wp_kernel<<<dim3(DIMn / 64, Bn / 128), 256, 0, stream>>>(abuf, wpb, y, bp, outp);

    gemm_mlp1_kernel<<<dim3(HIDn / 64, Bn / 128), 256, 0, stream>>>(yn2, w1b, bm1, hbuf);

    gemm_mlp2_kernel<<<dim3(DIMn / 64, Bn / 128), 256, 0, stream>>>(hbuf, w2b, bm2, outp);
}
